// Attention_50646254355249
// MI455X (gfx1250) — matrix operand order repaired and checked by reading, not compiled, not run on hardware
//
#include <hip/hip_runtime.h>
#include <hip/hip_bf16.h>

#define B_   2
#define T_   2048
#define H_   2048
#define NH_  16
#define NKV_ 4
#define HD_  128
#define GRP_ (NH_ / NKV_)            // 4
#define MTOT (B_ * T_)               // 4096
#define QKVN (NH_*HD_ + 2*NKV_*HD_)  // 3072

typedef __bf16 bf16;
typedef __attribute__((ext_vector_type(16))) __bf16 v16bf;
typedef __attribute__((ext_vector_type(8)))  float  v8f;

// ---- CDNA5 async-to-LDS path via inline asm (ISA §15.18.3 op 98) ----------
#if defined(__gfx1250__)
#  define HAVE_ASYNC 1
#else
#  define HAVE_ASYNC 0
#endif

#define LDS_AS __attribute__((address_space(3)))

__device__ __forceinline__ void copy16_g2l(const bf16* g, bf16* l) {
#if HAVE_ASYNC
  // GLOBAL_LOAD_ASYNC_TO_LDS_B128: VDST = per-lane LDS byte address,
  // VADDR = 64-bit global address, GV mode (no SADDR). Tracked by ASYNCcnt.
  const unsigned lofs = (unsigned)(size_t)(LDS_AS bf16*)l;
  asm volatile("global_load_async_to_lds_b128 %0, %1, off"
               :: "v"(lofs), "v"(g) : "memory");
#else
  *(uint4*)l = *(const uint4*)g;
#endif
}
__device__ __forceinline__ void wait_async_all() {
#if HAVE_ASYNC
  asm volatile("s_wait_asynccnt 0x0" ::: "memory");
#endif
}

// ---- bf16 conversion (RNE) -------------------------------------------------
__device__ __forceinline__ unsigned rne16(float f) {
  union { float f; unsigned u; } v; v.f = f;
  return (v.u + 0x7FFFu + ((v.u >> 16) & 1u)) >> 16;
}
__device__ __forceinline__ unsigned packbf2(float lo, float hi) {
  return rne16(lo) | (rne16(hi) << 16);
}
__device__ __forceinline__ bf16 f2bf(float f) {
  unsigned short h = (unsigned short)rne16(f);
  return __builtin_bit_cast(bf16, h);
}

// ---- Fragment loaders (wave32): contiguous k = 16*(lane/16) + s, same for A and B ------------------------

// A 16x32 bf16: lanes 0-15 row M=lane, K=0..15; lanes 16-31 K=16..31.
template <int LDK>
__device__ __forceinline__ v16bf load_a_frag(const bf16* tile, int m0, int k0) {
  const int lane = threadIdx.x & 31;
  const int row  = m0 + (lane & 15);
  const int kb   = k0 + ((lane >> 4) << 4);
  v16bf a;
#pragma unroll
  for (int i = 0; i < 8; ++i) {
    const int k = kb + 2*i;
    a[2*i]   = tile[row * LDK + k];
    a[2*i+1] = tile[row * LDK + k + 1];
  }
  return a;
}

// B 32x16 bf16 from [N][K] row-major tile (tile[n][k] = B(k,n)).
template <int LDK>
__device__ __forceinline__ v16bf load_b_frag_nk(const bf16* tile, int n0, int k0) {
  const int lane = threadIdx.x & 31;
  const int n  = n0 + (lane & 15);
  const int kb = k0 + ((lane >> 4) << 4);
  v16bf b;
#pragma unroll
  for (int i = 0; i < 8; ++i) {
    b[2*i]   = tile[n * LDK + kb + 2*i];
    b[2*i+1] = tile[n * LDK + kb + 2*i + 1];
  }
  return b;
}

// B 32x16 bf16 from [K][N] row-major tile (tile[k][n] = B(k,n)).
template <int LDN>
__device__ __forceinline__ v16bf load_b_frag_kn(const bf16* tile, int k0, int n0) {
  const int lane = threadIdx.x & 31;
  const int n  = n0 + (lane & 15);
  const int kb = k0 + ((lane >> 4) << 4);
  v16bf b;
#pragma unroll
  for (int i = 0; i < 8; ++i) {
    b[2*i]   = tile[(kb + 2*i    ) * LDN + n];
    b[2*i+1] = tile[(kb + 2*i + 1) * LDN + n];
  }
  return b;
}

// ---- Kernel 0: one-shot fp32 -> bf16 conversion ----------------------------

__global__ __launch_bounds__(256) void cvt_f32_bf16(
    const float* __restrict__ s, bf16* __restrict__ d, int n)
{
  const int i = (blockIdx.x * 256 + threadIdx.x) * 8;
  if (i >= n) return;
  const float4 a = *(const float4*)(s + i);
  const float4 b = *(const float4*)(s + i + 4);
  uint4 o;
  o.x = packbf2(a.x, a.y); o.y = packbf2(a.z, a.w);
  o.z = packbf2(b.x, b.y); o.w = packbf2(b.z, b.w);
  *(uint4*)(d + i) = o;
}

// ---- Kernel 1: C[M, ldc](+col0) = A[M,K] * W[N,K]^T, bf16 in, WMMA ---------

#define GT_M 128
#define GT_N 128
#define GT_K 32
#define LDK2 40   // 32 + 8: 80B row pitch (16B aligned), bank-spread

__global__ __launch_bounds__(256) void gemm_bf16wmma(
    const bf16* __restrict__ A, const bf16* __restrict__ W,
    float* __restrict__ C, int K, int ldc, int col0)
{
  __shared__ bf16 As[2][GT_M * LDK2];
  __shared__ bf16 Ws[2][GT_N * LDK2];

  const int tid  = threadIdx.x;
  const int wave = tid >> 5;
  const int lane = tid & 31;
  const int m0   = blockIdx.y * GT_M;
  const int n0   = blockIdx.x * GT_N;
  const int wm   = wave & 3;    // 32-row strip
  const int wn   = wave >> 2;   // 64-col strip

  v8f acc[2][4];
#pragma unroll
  for (int i = 0; i < 2; ++i)
#pragma unroll
    for (int j = 0; j < 4; ++j)
#pragma unroll
      for (int r = 0; r < 8; ++r) acc[i][j][r] = 0.0f;

  // stage one 128x32 tile of A and W into LDS buffer `buf`
  auto stage = [&](int buf, int kt) {
#pragma unroll
    for (int j = 0; j < 2; ++j) {
      const int idx = tid * 2 + j;          // 0..511 octs per matrix
      const int row = idx >> 2;
      const int oc  = (idx & 3) << 3;
      copy16_g2l(A + (size_t)(m0 + row) * K + kt + oc, &As[buf][row * LDK2 + oc]);
      copy16_g2l(W + (size_t)(n0 + row) * K + kt + oc, &Ws[buf][row * LDK2 + oc]);
    }
  };
  auto compute = [&](int buf) {
    v16bf af[2], bw[4];
#pragma unroll
    for (int i = 0; i < 2; ++i) af[i] = load_a_frag<LDK2>(As[buf], wm * 32 + i * 16, 0);
#pragma unroll
    for (int j = 0; j < 4; ++j) bw[j] = load_b_frag_nk<LDK2>(Ws[buf], wn * 64 + j * 16, 0);
#pragma unroll
    for (int i = 0; i < 2; ++i)
#pragma unroll
      for (int j = 0; j < 4; ++j)
        acc[i][j] = __builtin_amdgcn_wmma_f32_16x16x32_bf16(
            false, af[i], false, bw[j], (short)0, acc[i][j], false, false);
  };

  const int niter = K / GT_K;
#if HAVE_ASYNC
  stage(0, 0);                                 // prologue
  for (int it = 0; it < niter; ++it) {
    wait_async_all();                          // my tile's async ops done
    __syncthreads();                           // everyone's tile visible / prev reads done
    if (it + 1 < niter) stage((it + 1) & 1, (it + 1) * GT_K);  // overlap with WMMA
    compute(it & 1);
  }
#else
  for (int it = 0; it < niter; ++it) {
    __syncthreads();
    stage(0, it * GT_K);
    __syncthreads();
    compute(0);
  }
#endif

  const int nloc = lane & 15;
  const int mhal = (lane >> 4) << 3;
#pragma unroll
  for (int i = 0; i < 2; ++i)
#pragma unroll
    for (int j = 0; j < 4; ++j) {
      const int n = col0 + n0 + wn * 64 + j * 16 + nloc;
#pragma unroll
      for (int r = 0; r < 8; ++r) {
        const int m = m0 + wm * 32 + i * 16 + mhal + r;
        C[(size_t)m * ldc + n] = acc[i][j][r];
      }
    }
}

// ---- Kernel 2: RMSNorm + RoPE + pack to [B, heads, T, HD] bf16 -------------

__global__ __launch_bounds__(128) void norm_rope_pack(
    const float* __restrict__ qkv, const float* __restrict__ qw,
    const float* __restrict__ kw,
    bf16* __restrict__ Qb, bf16* __restrict__ Kb, bf16* __restrict__ Vb)
{
  const int lane  = threadIdx.x & 31;
  const int gw    = blockIdx.x * 4 + (threadIdx.x >> 5);
  const int token = gw / (NH_ + 2 * NKV_);
  const int hh    = gw % (NH_ + 2 * NKV_);
  const int b = token / T_;
  const int t = token % T_;

  const float* row = qkv + (size_t)token * QKVN;
  int col0, kind;
  bf16* outp;
  if (hh < NH_) {
    kind = 0; col0 = hh * HD_;
    outp = Qb + (((size_t)(b * NH_ + hh)) * T_ + t) * HD_;
  } else if (hh < NH_ + NKV_) {
    kind = 1; col0 = NH_ * HD_ + (hh - NH_) * HD_;
    outp = Kb + (((size_t)(b * NKV_ + (hh - NH_))) * T_ + t) * HD_;
  } else {
    kind = 2; col0 = (NH_ + NKV_) * HD_ + (hh - NH_ - NKV_) * HD_;
    outp = Vb + (((size_t)(b * NKV_ + (hh - NH_ - NKV_))) * T_ + t) * HD_;
  }

  const int j0 = lane * 2;                 // rope pair (j0, j0+64)
  float f0 = row[col0 + j0];
  float f1 = row[col0 + j0 + 1];
  float g0 = row[col0 + j0 + 64];
  float g1 = row[col0 + j0 + 65];

  if (kind == 2) {                          // V: convert only
    outp[j0] = f2bf(f0); outp[j0+1] = f2bf(f1);
    outp[j0+64] = f2bf(g0); outp[j0+65] = f2bf(g1);
    return;
  }
  float ss = f0*f0 + f1*f1 + g0*g0 + g1*g1;
#pragma unroll
  for (int m = 1; m < 32; m <<= 1) ss += __shfl_xor(ss, m, 32);
  const float rstd = rsqrtf(ss * (1.0f / HD_) + 1e-6f);
  const float* w = (kind == 0) ? qw : kw;
  const float n0v = f0 * rstd * (1.0f + w[j0]);
  const float n1v = f1 * rstd * (1.0f + w[j0 + 1]);
  const float m0v = g0 * rstd * (1.0f + w[j0 + 64]);
  const float m1v = g1 * rstd * (1.0f + w[j0 + 65]);

  const float inv0 = __powf(10000.0f, -(float)j0 / 64.0f);
  const float inv1 = __powf(10000.0f, -(float)(j0 + 1) / 64.0f);
  float s0, c0, s1, c1;
  __sincosf((float)t * inv0, &s0, &c0);
  __sincosf((float)t * inv1, &s1, &c1);

  outp[j0]      = f2bf(n0v * c0 - m0v * s0);
  outp[j0 + 1]  = f2bf(n1v * c1 - m1v * s1);
  outp[j0 + 64] = f2bf(m0v * c0 + n0v * s0);
  outp[j0 + 65] = f2bf(m1v * c1 + n1v * s1);
}

// ---- Kernel 3: causal flash attention (GQA), bf16 WMMA, online softmax -----

#define ATQ 64
#define ATK 32
#define LDH 136   // HD_+8: 272B row pitch, 16B aligned
#define LDP 34

__global__ __launch_bounds__(128) void flash_attn(
    const bf16* __restrict__ Qb, const bf16* __restrict__ Kb,
    const bf16* __restrict__ Vb, bf16* __restrict__ O)
{
  __shared__ bf16 Qs[ATQ * LDH];
  __shared__ bf16 Ks[ATK * LDH];
  __shared__ bf16 Vs[ATK * LDH];
  __shared__ bf16 Ps[4 * 16 * LDP];

  const int tid  = threadIdx.x;
  const int wave = tid >> 5;
  const int lane = tid & 31;
  const int h   = blockIdx.y;
  const int b   = blockIdx.z;
  const int q0  = blockIdx.x * ATQ;
  const int kvh = h / GRP_;

  const bf16* Qg = Qb + ((size_t)(b * NH_  + h  )) * T_ * HD_;
  const bf16* Kg = Kb + ((size_t)(b * NKV_ + kvh)) * T_ * HD_;
  const bf16* Vg = Vb + ((size_t)(b * NKV_ + kvh)) * T_ * HD_;

  // Stage Q tile (64x128) once: 1024 b128 chunks / 128 threads = 8 each.
  for (int i = tid; i < ATQ * (HD_ / 8); i += 128) {
    const int r = i >> 4;
    const int c = (i & 15) << 3;
    copy16_g2l(Qg + (size_t)(q0 + r) * HD_ + c, Qs + r * LDH + c);
  }
  wait_async_all();
  __syncthreads();

  v16bf qf[4];
#pragma unroll
  for (int kk = 0; kk < 4; ++kk) qf[kk] = load_a_frag<LDH>(Qs, wave * 16, kk * 32);

  v8f o[8];
#pragma unroll
  for (int i = 0; i < 8; ++i)
#pragma unroll
    for (int r = 0; r < 8; ++r) o[i][r] = 0.0f;
  float rm[8], rl[8];
#pragma unroll
  for (int r = 0; r < 8; ++r) { rm[r] = -3.0e38f; rl[r] = 0.0f; }

  const float scale = 0.08838834764831845f;   // 1/sqrt(128)
  const int qend = q0 + ATQ - 1;
  const int mhal = (lane >> 4) << 3;
  const int nloc = lane & 15;

  for (int kt = 0; kt <= qend; kt += ATK) {
    __syncthreads();                                   // prior tile reads done
    for (int i = tid; i < ATK * (HD_ / 8); i += 128) { // 512 chunks -> 4 each/matrix
      const int r = i >> 4;
      const int c = (i & 15) << 3;
      copy16_g2l(Kg + (size_t)(kt + r) * HD_ + c, Ks + r * LDH + c);
      copy16_g2l(Vg + (size_t)(kt + r) * HD_ + c, Vs + r * LDH + c);
    }
    wait_async_all();
    __syncthreads();

    // S = Q * K^T  (two 16-key subtiles, HD=128 -> 4 chained WMMAs each)
    v8f s[2];
#pragma unroll
    for (int j = 0; j < 2; ++j)
#pragma unroll
      for (int r = 0; r < 8; ++r) s[j][r] = 0.0f;
#pragma unroll
    for (int kk = 0; kk < 4; ++kk)
#pragma unroll
      for (int j = 0; j < 2; ++j) {
        v16bf bfK = load_b_frag_nk<LDH>(Ks, j * 16, kk * 32);
        s[j] = __builtin_amdgcn_wmma_f32_16x16x32_bf16(
            false, qf[kk], false, bfK, (short)0, s[j], false, false);
      }

    // scale + causal mask
#pragma unroll
    for (int j = 0; j < 2; ++j) {
      const int kcol = kt + j * 16 + nloc;
#pragma unroll
      for (int r = 0; r < 8; ++r) {
        const int qrow = q0 + wave * 16 + mhal + r;
        const float v = s[j][r] * scale;
        s[j][r] = (kcol <= qrow) ? v : -3.0e38f;
      }
    }

    // online softmax per row (rows live on 16-lane halves -> xor 1,2,4,8)
#pragma unroll
    for (int r = 0; r < 8; ++r) {
      float mx = fmaxf(s[0][r], s[1][r]);
#pragma unroll
      for (int msk = 1; msk < 16; msk <<= 1) mx = fmaxf(mx, __shfl_xor(mx, msk, 32));
      const float nm = fmaxf(rm[r], mx);
      const float cf = __expf(rm[r] - nm);
      rm[r] = nm;
      rl[r] *= cf;
#pragma unroll
      for (int i = 0; i < 8; ++i) o[i][r] *= cf;
      const float p0 = __expf(s[0][r] - nm);
      const float p1 = __expf(s[1][r] - nm);
      s[0][r] = p0; s[1][r] = p1;
      float ps = p0 + p1;
#pragma unroll
      for (int msk = 1; msk < 16; msk <<= 1) ps += __shfl_xor(ps, msk, 32);
      rl[r] += ps;
    }

    // C-layout -> A-layout via wave-private LDS staging (DS in-order per wave)
    bf16* Pw = Ps + wave * 16 * LDP;
#pragma unroll
    for (int j = 0; j < 2; ++j)
#pragma unroll
      for (int r = 0; r < 8; ++r)
        Pw[(mhal + r) * LDP + j * 16 + nloc] = f2bf(s[j][r]);
    const v16bf pf = load_a_frag<LDP>(Pw, 0, 0);

    // O += P * V  (K-dim = 32 keys, 8 output subtiles across HD)
#pragma unroll
    for (int i = 0; i < 8; ++i) {
      v16bf bfV = load_b_frag_kn<LDH>(Vs, 0, i * 16);
      o[i] = __builtin_amdgcn_wmma_f32_16x16x32_bf16(
          false, pf, false, bfV, (short)0, o[i], false, false);
    }
  }

  // epilogue: O / l -> bf16 [B,T,NH*HD] for the out-projection GEMM
#pragma unroll
  for (int i = 0; i < 8; ++i) {
    const int d = i * 16 + nloc;
#pragma unroll
    for (int r = 0; r < 8; ++r) {
      const int q = q0 + wave * 16 + mhal + r;
      O[((size_t)(b * T_ + q)) * (NH_ * HD_) + h * HD_ + d] = f2bf(o[i][r] / rl[r]);
    }
  }
}

// ---- Host-side orchestration ----------------------------------------------

extern "C" void kernel_launch(void* const* d_in, const int* in_sizes, int n_in,
                              void* d_out, int out_size, void* d_ws, size_t ws_size,
                              hipStream_t stream) {
  const float* x  = (const float*)d_in[0];
  const float* Wq = (const float*)d_in[1];
  const float* Wk = (const float*)d_in[2];
  const float* Wv = (const float*)d_in[3];
  const float* Wo = (const float*)d_in[4];
  const float* qw = (const float*)d_in[5];
  const float* kw = (const float*)d_in[6];
  float* out = (float*)d_out;

  char* ws = (char*)d_ws;
  float* qkv = (float*)ws;                       // [4096,3072] f32 (reused as attnb)
  size_t off = (size_t)MTOT * QKVN * sizeof(float);
  bf16* xb  = (bf16*)(ws + off); off += (size_t)MTOT * H_          * sizeof(bf16);
  bf16* Wqb = (bf16*)(ws + off); off += (size_t)NH_  * HD_ * H_    * sizeof(bf16);
  bf16* Wkb = (bf16*)(ws + off); off += (size_t)NKV_ * HD_ * H_    * sizeof(bf16);
  bf16* Wvb = (bf16*)(ws + off); off += (size_t)NKV_ * HD_ * H_    * sizeof(bf16);
  bf16* Wob = (bf16*)(ws + off); off += (size_t)H_ * NH_ * HD_     * sizeof(bf16);
  bf16* Qb  = (bf16*)(ws + off); off += (size_t)B_ * NH_  * T_ * HD_ * sizeof(bf16);
  bf16* Kb  = (bf16*)(ws + off); off += (size_t)B_ * NKV_ * T_ * HD_ * sizeof(bf16);
  bf16* Vb  = (bf16*)(ws + off); off += (size_t)B_ * NKV_ * T_ * HD_ * sizeof(bf16);
  bf16* attnb = (bf16*)qkv;                      // QKV dead after norm_rope_pack

  // One-shot bf16 conversion of activations + weights
  auto cvt = [&](const float* s, bf16* d, int n) {
    cvt_f32_bf16<<<n / 2048, 256, 0, stream>>>(s, d, n);
  };
  cvt(x,  xb,  MTOT * H_);
  cvt(Wq, Wqb, NH_  * HD_ * H_);
  cvt(Wk, Wkb, NKV_ * HD_ * H_);
  cvt(Wv, Wvb, NKV_ * HD_ * H_);
  cvt(Wo, Wob, H_ * NH_ * HD_);

  // Fused QKV projection (three column slabs of one [4096,3072] matrix)
  gemm_bf16wmma<<<dim3((NH_*HD_)/GT_N, MTOT/GT_M), 256, 0, stream>>>(
      xb, Wqb, qkv, H_, QKVN, 0);
  gemm_bf16wmma<<<dim3((NKV_*HD_)/GT_N, MTOT/GT_M), 256, 0, stream>>>(
      xb, Wkb, qkv, H_, QKVN, NH_ * HD_);
  gemm_bf16wmma<<<dim3((NKV_*HD_)/GT_N, MTOT/GT_M), 256, 0, stream>>>(
      xb, Wvb, qkv, H_, QKVN, (NH_ + NKV_) * HD_);

  norm_rope_pack<<<(MTOT * (NH_ + 2 * NKV_)) / 4, 128, 0, stream>>>(
      qkv, qw, kw, Qb, Kb, Vb);

  flash_attn<<<dim3(T_ / ATQ, NH_, B_), 128, 0, stream>>>(Qb, Kb, Vb, attnb);

  gemm_bf16wmma<<<dim3(H_/GT_N, MTOT/GT_M), 256, 0, stream>>>(
      attnb, Wob, out, NH_ * HD_, H_, 0);
}
